// MSCNDANN_60842506715653
// MI455X (gfx1250) — matrix operand order repaired and checked by reading, not compiled, not run on hardware
//
#include <hip/hip_runtime.h>
#include <hip/hip_bf16.h>
#include <stdint.h>

// ---------------------------------------------------------------------------
// MeshNet-style point-cloud net for MI455X (gfx1250).
//  * GEMMs  -> v_wmma_f32_16x16x32_f16 (f32 accum); A tiles staged with
//              global_load_async_to_lds_b128, W tiles with tensor_load_to_lds,
//              double-buffered with threshold waits (asynccnt<=4, tensorcnt<=1)
//              so the TDM/async engines run ahead of the WMMA pipe.
//  * KNN    -> v_wmma_f32_16x16x4_f32 for the (x,y,z,0) inner products.
//  * Working set (~35MB) fits in the 192MB L2; at 23.3 TB/s the net is
//    compute/latency bound, so matrix pipes + async staging are the win.
// ---------------------------------------------------------------------------

#define S_KER   4
#define KNN_K   20
#define EPS_NRM 1e-12f
#define BN_EPS_F 1e-05f
#define KCAP    21

typedef __attribute__((ext_vector_type(2)))  float    v2f;
typedef __attribute__((ext_vector_type(8)))  float    v8f;
typedef __attribute__((ext_vector_type(16))) _Float16 v16h;
typedef __attribute__((ext_vector_type(4)))  unsigned v4u;
typedef __attribute__((ext_vector_type(8)))  int      v8i;
typedef __attribute__((ext_vector_type(4)))  int      v4i;

#define CDIV(a, b) (((a) + (b) - 1) / (b))

// gfx1250: flat shared addresses keep the LDS byte offset in the low 32 bits
// (ISA 10.2).
__device__ __forceinline__ unsigned lds_addr32(const void* p) {
  return (unsigned)(unsigned long long)p;
}

__device__ __forceinline__ void async_load_b128(unsigned lds_byte, const void* gptr) {
  unsigned long long ga = (unsigned long long)gptr;
  asm volatile("global_load_async_to_lds_b128 %0, %1, off"
               :: "v"(lds_byte), "v"(ga) : "memory");
}
template <int N>
__device__ __forceinline__ void wait_async_le() {
  asm volatile("s_wait_asynccnt %0" :: "i"(N) : "memory");
}
__device__ __forceinline__ void wait_dscnt0() {
  asm volatile("s_wait_dscnt 0" ::: "memory");
}

// TDM: 2D f32 tile (tile_d1 rows x tile_d0 cols) from row-major tensor
// (stride_d0 elements per row) into contiguous LDS. D# layout per ISA ch.8.
// 6-arg builtin: (v4u g0, v8i g1, v4i g2, v4i g3, v8i g4, i32 cpol)
__device__ __forceinline__ void tdm_load_2d_f32(unsigned lds_byte, const float* gsrc,
                                                unsigned tensor_d0, unsigned tensor_d1,
                                                unsigned tile_d0, unsigned tile_d1,
                                                unsigned stride_d0) {
  unsigned long long ga = (unsigned long long)(const void*)gsrc;
  v4u g0;
  g0[0] = 1u;                                   // count=1, user descriptor
  g0[1] = lds_byte;                             // lds_addr
  g0[2] = (unsigned)(ga & 0xffffffffu);         // global_addr[31:0]
  g0[3] = (unsigned)((ga >> 32) & 0x1ffffffu) | (2u << 30); // addr[56:32], type=2
  v8i g1;
  g1[0] = (int)(2u << 16);                      // data_size=4B, workgroup_mask=0
  g1[1] = (int)((tensor_d0 & 0xffffu) << 16);   // tensor_dim0 lo
  g1[2] = (int)((tensor_d0 >> 16) | ((tensor_d1 & 0xffffu) << 16));
  g1[3] = (int)((tensor_d1 >> 16) | ((tile_d0 & 0xffffu) << 16));
  g1[4] = (int)(tile_d1 & 0xffffu);             // tile_dim1, tile_dim2=0
  g1[5] = (int)stride_d0;                       // tensor_dim0_stride lo
  g1[6] = 0;
  g1[7] = 0;
  v4i z4 = {0, 0, 0, 0};
  v8i z8 = {0, 0, 0, 0, 0, 0, 0, 0};
  __builtin_amdgcn_tensor_load_to_lds(g0, g1, z4, z4, z8, 0);
}

// Build f16 fragments from staged f32 LDS tiles and run one WMMA step.
// A tile: 16x32 row-major; B tile: 32x16 row-major (ISA 7.12.2 layouts).
__device__ __forceinline__ v8f wmma_step_f16(const float (*a)[32], const float (*bm)[16],
                                             int half, int l16, v8f acc) {
  v16h afrag, bfrag;
#pragma unroll
  for (int v = 0; v < 8; ++v) {
    int ka = 2 * v + (v >= 4 ? 8 : 0) + 8 * half;
    afrag[2 * v + 0] = (_Float16)a[l16][ka + 0];
    afrag[2 * v + 1] = (_Float16)a[l16][ka + 1];
    int kb = 2 * v + (v >= 4 ? 8 : 0) + 8 * half;
    bfrag[2 * v + 0] = (_Float16)bm[kb + 0][l16];
    bfrag[2 * v + 1] = (_Float16)bm[kb + 1][l16];
  }
  return __builtin_amdgcn_wmma_f32_16x16x32_f16(false, afrag, false, bfrag,
                                                (short)0, acc, false, false);
}

// ---------------------------------------------------------------------------
// sd = dir / max(||dir||_col, eps), dir is (3, Kc) row-major
// ---------------------------------------------------------------------------
__global__ void normalize_cols(const float* __restrict__ dir, float* __restrict__ sd,
                               int Kc) {
  int k = blockIdx.x * blockDim.x + threadIdx.x;
  if (k >= Kc) return;
  float a = dir[k], b = dir[Kc + k], c = dir[2 * Kc + k];
  float inv = 1.f / fmaxf(sqrtf(a * a + b * b + c * c), EPS_NRM);
  sd[k] = a * inv; sd[Kc + k] = b * inv; sd[2 * Kc + k] = c * inv;
}

// ---------------------------------------------------------------------------
// KNN: block = 16 query rows, 8 waves sweep column tiles with f32 WMMA.
// dist = |a|^2 + |b|^2 - 2 a.b ; per-lane sorted top-(k+1) then 16-way merge.
// Writes k neighbors (self excluded) per row. Requires V % 128 == 0.
// ---------------------------------------------------------------------------
__global__ void knn_topk(const float* __restrict__ verts, int* __restrict__ idx_out,
                         int V, int kk /* = k+1 */) {
  __shared__ float rowv[16][4];
  __shared__ float rowq[16];
  __shared__ float dtile[8][16][17];
  __shared__ float candd[8][32][KCAP];
  __shared__ int   candi[8][32][KCAP];

  const int b = blockIdx.y;
  const int mt = blockIdx.x;
  const int tid = (int)threadIdx.x;
  const int w = tid >> 5, L = tid & 31;
  const int half = L >> 4, l16 = L & 15;

  if (tid < 16) {
    int r = mt * 16 + tid;
    float x = verts[((size_t)b * V + r) * 3 + 0];
    float y = verts[((size_t)b * V + r) * 3 + 1];
    float z = verts[((size_t)b * V + r) * 3 + 2];
    rowv[tid][0] = x; rowv[tid][1] = y; rowv[tid][2] = z; rowv[tid][3] = 0.f;
    rowq[tid] = x * x + y * y + z * z;
  }
  __syncthreads();

  float ld[KCAP]; int li[KCAP];
  for (int t = 0; t < kk; ++t) { ld[t] = 3.0e38f; li[t] = -1; }

  const int ntiles = V >> 4;                 // divisible by 8 for V in {4096,1024,256}
  for (int ct = w; ct < ntiles; ct += 8) {
    const int col = ct * 16 + l16;
    const float cx = verts[((size_t)b * V + col) * 3 + 0];
    const float cy = verts[((size_t)b * V + col) * 3 + 1];
    const float cz = verts[((size_t)b * V + col) * 3 + 2];
    const float qc = cx * cx + cy * cy + cz * cz;

    v2f af, bf;
    af[0] = (half == 0) ? rowv[l16][0] : rowv[l16][2];
    af[1] = (half == 0) ? rowv[l16][1] : rowv[l16][3];
    bf[0] = (half == 0) ? cx : cz;
    bf[1] = (half == 0) ? cy : 0.f;

    v8f cacc = {};
    cacc = __builtin_amdgcn_wmma_f32_16x16x4_f32(false, af, false, bf,
                                                 (short)0, cacc, false, false);
#pragma unroll
    for (int v = 0; v < 8; ++v) {
      int m = v + 8 * half;
      dtile[w][m][l16] = rowq[m] + qc - 2.0f * cacc[v];
    }
    __syncthreads();

    const int cb = half * 8;
    for (int j = 0; j < 8; ++j) {
      float d = dtile[w][l16][cb + j];
      int ci = ct * 16 + cb + j;
      if (d < ld[kk - 1]) {
        int p = kk - 1;
        while (p > 0 && ld[p - 1] > d) { ld[p] = ld[p - 1]; li[p] = li[p - 1]; --p; }
        ld[p] = d; li[p] = ci;
      }
    }
    __syncthreads();
  }

  for (int t = 0; t < kk; ++t) { candd[w][L][t] = ld[t]; candi[w][L][t] = li[t]; }
  __syncthreads();

  if (tid < 16) {                             // merge 16 sorted lists per row
    int hp[16];
    for (int l = 0; l < 16; ++l) hp[l] = 0;
    const int row = mt * 16 + tid;
    const int k = kk - 1;
    for (int s = 0; s < kk; ++s) {
      float best = __FLT_MAX__; int bl = 0, bi = -1;
      for (int l = 0; l < 16; ++l) {
        int p = hp[l];
        if (p < kk) {
          int lane = tid + ((l & 1) << 4);
          float vv = candd[l >> 1][lane][p];
          if (vv < best) { best = vv; bl = l; bi = candi[l >> 1][lane][p]; }
        }
      }
      hp[bl]++;
      if (s > 0) idx_out[((size_t)b * V + row) * k + (s - 1)] = bi;  // drop self
    }
  }
}

// ---------------------------------------------------------------------------
// normalized neighbor directions: dirn[(bv*k+n)*3+i]
// ---------------------------------------------------------------------------
__global__ void nbr_dirs(const float* __restrict__ verts, const int* __restrict__ idx,
                         float* __restrict__ dirn, int B, int V, int k) {
  int t = blockIdx.x * blockDim.x + threadIdx.x;
  int total = B * V * k;
  if (t >= total) return;
  int v = (t / k) % V;
  int b = t / (k * V);
  int j = idx[t];
  float dx = verts[((size_t)b * V + j) * 3 + 0] - verts[((size_t)b * V + v) * 3 + 0];
  float dy = verts[((size_t)b * V + j) * 3 + 1] - verts[((size_t)b * V + v) * 3 + 1];
  float dz = verts[((size_t)b * V + j) * 3 + 2] - verts[((size_t)b * V + v) * 3 + 2];
  float inv = 1.f / fmaxf(sqrtf(dx * dx + dy * dy + dz * dz), EPS_NRM);
  dirn[(size_t)t * 3 + 0] = dx * inv;
  dirn[(size_t)t * 3 + 1] = dy * inv;
  dirn[(size_t)t * 3 + 2] = dz * inv;
}

// ---------------------------------------------------------------------------
// GEMM  C[R,N] = A[R,K] @ W[K,N] + bias, f16 WMMA with f32 accumulate.
// One wave per 16x16 C tile. Double-buffered LDS staging: A via async
// global->LDS copies, W via TDM; threshold waits keep the next stage's
// 4 async loads + 1 tensor op in flight while the current stage computes.
// ---------------------------------------------------------------------------
__global__ void gemm_wmma_f16(const float* __restrict__ A, const float* __restrict__ W,
                              const float* __restrict__ bias, float* __restrict__ C,
                              int R, int K, int N) {
  __shared__ __align__(16) float aT[8][2][16][32];
  __shared__ __align__(16) float bT[8][2][32][16];
  const int w = (int)threadIdx.x >> 5;
  const int L = (int)threadIdx.x & 31;
  const int half = L >> 4, l16 = L & 15;
  const int ntiles = N >> 4;
  const int nt = blockIdx.x * 8 + w;
  if (nt >= ntiles) return;                    // wave-uniform; no block barriers used
  const int m0 = (int)blockIdx.y << 4;
  const int n0 = nt << 4;
  v8f acc = {};

  if ((K & 31) == 0) {
    // ---- pipelined fast path -------------------------------------------
    auto stage = [&](int k0, int buf) {
#pragma unroll
      for (int i = 0; i < 4; ++i) {
        int e = (i * 32 + L) << 2;            // float index within 16x32 tile
        int tr = e >> 5, tc = e & 31;
        int row = m0 + tr;
        if (row >= R) row = R - 1;            // clamp: always exactly 4 loads/stage
        async_load_b128(lds_addr32(&aT[w][buf][tr][tc]),
                        A + (size_t)row * K + (k0 + tc));
      }
      tdm_load_2d_f32(lds_addr32(&bT[w][buf][0][0]), W + (size_t)k0 * N + n0,
                      (unsigned)N, (unsigned)K, 16u, 32u, (unsigned)N);
    };
    const int nsteps = K >> 5;
    int buf = 0;
    stage(0, 0);
    for (int s = 0; s < nsteps; ++s) {
      if (s + 1 < nsteps) {
        stage((s + 1) << 5, buf ^ 1);
        __builtin_prefetch(A + (size_t)m0 * K + ((s + 2) << 5), 0, 1);
        wait_async_le<4>();                   // stage s done; s+1 still in flight
        __builtin_amdgcn_s_wait_tensorcnt(1);
      } else {
        wait_async_le<0>();
        __builtin_amdgcn_s_wait_tensorcnt(0);
      }
      acc = wmma_step_f16(aT[w][buf], bT[w][buf], half, l16, acc);
      buf ^= 1;
    }
  } else {
    // ---- K==16 tail path: scalar stage with zero padding ----------------
    for (int k0 = 0; k0 < K; k0 += 32) {
      for (int e = L; e < 512; e += 32) {
        int tr = e >> 5, tc = e & 31;
        aT[w][0][tr][tc] = ((k0 + tc) < K && (m0 + tr) < R)
                               ? A[(size_t)(m0 + tr) * K + (k0 + tc)] : 0.f;
        int kr = e >> 4, nc = e & 15;
        bT[w][0][kr][nc] = ((k0 + kr) < K) ? W[(size_t)(k0 + kr) * N + (n0 + nc)] : 0.f;
      }
      wait_dscnt0();
      acc = wmma_step_f16(aT[w][0], bT[w][0], half, l16, acc);
    }
  }

  const float bv = bias[n0 + l16];
#pragma unroll
  for (int v = 0; v < 8; ++v) {
    int m = m0 + v + 8 * half;
    if (m < R) C[(size_t)m * N + (n0 + l16)] = acc[v] + bv;
  }
}

// ---------------------------------------------------------------------------
// fused conv: out[b,v,c] = relu(center + sum_s max_n relu(dirn.sd) * support)
// feat == nullptr -> conv_surface (theta only, no center/support)
// ---------------------------------------------------------------------------
__global__ void conv_theta(const float* __restrict__ feat, const float* __restrict__ dirn,
                           const int* __restrict__ idx, const float* __restrict__ sd,
                           float* __restrict__ out, int B, int V, int outc, int knn) {
  int t = blockIdx.x * blockDim.x + threadIdx.x;
  int total = B * V * outc;
  if (t >= total) return;
  int c = t % outc;
  int v = (t / outc) % V;
  int b = t / (outc * V);
  const int Ktot = S_KER * outc;
  const int Ncols = (S_KER + 1) * outc;
  const size_t bv = (size_t)b * V + v;
  float acc = feat ? feat[bv * Ncols + c] : 0.f;
  for (int s = 0; s < S_KER; ++s) {
    int scol = s * outc + c;
    float s0 = sd[scol], s1 = sd[Ktot + scol], s2 = sd[2 * Ktot + scol];
    float mx = -3.0e38f;
    for (int n = 0; n < knn; ++n) {
      size_t db = (bv * knn + n) * 3;
      float th = dirn[db] * s0 + dirn[db + 1] * s1 + dirn[db + 2] * s2;
      th = th > 0.f ? th : 0.f;
      float val;
      if (feat) {
        int j = idx[bv * knn + n];
        val = th * feat[((size_t)b * V + j) * Ncols + outc + scol];
      } else {
        val = th;
      }
      mx = fmaxf(mx, val);
    }
    acc += mx;
  }
  out[bv * outc + c] = fmaxf(acc, 0.f);
}

// ---------------------------------------------------------------------------
// pooling helpers (deterministic first-V/4 sample in place of jax permutation)
// ---------------------------------------------------------------------------
__global__ void pool_verts(const float* __restrict__ vin, float* __restrict__ vout,
                           int Vin, int Vout, int B) {
  int t = blockIdx.x * blockDim.x + threadIdx.x;
  int total = B * Vout * 3;
  if (t >= total) return;
  int b = t / (Vout * 3);
  int r = t % (Vout * 3);
  vout[(size_t)b * Vout * 3 + r] = vin[(size_t)b * Vin * 3 + r];
}

__global__ void pool_max(const float* __restrict__ fm, const int* __restrict__ idx,
                         float* __restrict__ out, int B, int Vin, int Vout, int C, int nbk) {
  int t = blockIdx.x * blockDim.x + threadIdx.x;
  int total = B * Vout * C;
  if (t >= total) return;
  int c = t % C;
  int vs = (t / C) % Vout;
  int b = t / (C * Vout);
  float m = -3.0e38f;
  for (int j = 0; j < nbk; ++j) {
    int jn = idx[((size_t)b * Vin + vs) * nbk + j];
    m = fmaxf(m, fm[((size_t)b * Vin + jn) * C + c]);
  }
  out[((size_t)b * Vout + vs) * C + c] = m;
}

__global__ void colmax(const float* __restrict__ fm, float* __restrict__ g,
                       int B, int V, int C) {
  int t = blockIdx.x * blockDim.x + threadIdx.x;
  if (t >= B * C) return;
  int b = t / C, c = t % C;
  float m = -3.0e38f;
  for (int v = 0; v < V; ++v) m = fmaxf(m, fm[((size_t)b * V + v) * C + c]);
  g[t] = m;
}

__global__ void concat_g(const float* __restrict__ fm, const float* __restrict__ g,
                         float* __restrict__ out, int B, int V, int C) {
  int t = blockIdx.x * blockDim.x + threadIdx.x;
  int total = B * V * 2 * C;
  if (t >= total) return;
  int c2 = t % (2 * C);
  int v = (t / (2 * C)) % V;
  int b = t / (2 * C * V);
  out[t] = (c2 < C) ? fm[((size_t)b * V + v) * C + c2] : g[b * C + (c2 - C)];
}

__global__ void bn_relu(float* __restrict__ x, const float* __restrict__ g,
                        const float* __restrict__ bt, int B, int C) {
  int t = blockIdx.x * blockDim.x + threadIdx.x;
  if (t >= B * C) return;
  int c = t % C;
  float v = x[t] * (g[c] * rsqrtf(1.f + BN_EPS_F)) + bt[c];
  x[t] = fmaxf(v, 0.f);
}

__global__ void head_out(const float* __restrict__ h, const float* __restrict__ w2,
                         const float* __restrict__ b2, float* __restrict__ out,
                         int B, int C, int ncls) {
  int b = (int)threadIdx.x;
  if (b >= B) return;
  float logit[4];
  for (int j = 0; j < ncls; ++j) {
    float a = b2[j];
    for (int c = 0; c < C; ++c) a += h[(size_t)b * C + c] * w2[(size_t)c * ncls + j];
    logit[j] = a;
  }
  float mx = logit[0];
  for (int j = 1; j < ncls; ++j) mx = fmaxf(mx, logit[j]);
  float se = 0.f;
  for (int j = 0; j < ncls; ++j) se += expf(logit[j] - mx);
  float lse = logf(se) + mx;
  for (int j = 0; j < ncls; ++j) out[(size_t)b * ncls + j] = logit[j] - lse;
}

// ---------------------------------------------------------------------------
extern "C" void kernel_launch(void* const* d_in, const int* in_sizes, int n_in,
                              void* d_out, int out_size, void* d_ws, size_t ws_size,
                              hipStream_t stream) {
  (void)in_sizes; (void)n_in; (void)out_size; (void)ws_size;

  const float* verts0 = (const float*)d_in[0];
  // d_in[1] = alpha (unused in forward values)
  const float* dir0 = (const float*)d_in[2];
  const float* w1   = (const float*)d_in[3];
  const float* b1   = (const float*)d_in[4];
  const float* dir1 = (const float*)d_in[5];
  const float* w2   = (const float*)d_in[6];
  const float* b2   = (const float*)d_in[7];
  const float* dir2 = (const float*)d_in[8];
  const float* w3   = (const float*)d_in[9];
  const float* b3   = (const float*)d_in[10];
  const float* dir3 = (const float*)d_in[11];
  const float* w4   = (const float*)d_in[12];
  const float* b4   = (const float*)d_in[13];
  const float* dir4 = (const float*)d_in[14];
  const float* w5   = (const float*)d_in[15];
  const float* b5   = (const float*)d_in[16];
  const float* dir5 = (const float*)d_in[17];
  const float* lc1w = (const float*)d_in[18];
  const float* lc1b = (const float*)d_in[19];
  const float* bncg = (const float*)d_in[20];
  const float* bncb = (const float*)d_in[21];
  const float* lc2w = (const float*)d_in[22];
  const float* lc2b = (const float*)d_in[23];
  const float* ld1w = (const float*)d_in[24];
  const float* ld1b = (const float*)d_in[25];
  const float* bndg = (const float*)d_in[26];
  const float* bndb = (const float*)d_in[27];
  const float* ld2w = (const float*)d_in[28];
  const float* ld2b = (const float*)d_in[29];

  const int B = 4, V0 = 4096, V1 = 1024, V2 = 256;

  char* ws = (char*)d_ws;
  size_t off = 0;
  auto alloc = [&](size_t bytes) -> void* {
    void* p = ws + off;
    off += (bytes + 255) & ~(size_t)255;
    return p;
  };

  float* sd0 = (float*)alloc(3 * 64 * 4);
  float* sd1 = (float*)alloc(3 * 128 * 4);
  float* sd2 = (float*)alloc(3 * 256 * 4);
  float* sd3 = (float*)alloc(3 * 1024 * 4);
  float* sd4 = (float*)alloc(3 * 1024 * 4);
  float* sd5 = (float*)alloc(3 * 4096 * 4);
  int*   idxb   = (int*)alloc((size_t)B * V0 * KNN_K * 4);
  float* dirn   = (float*)alloc((size_t)B * V0 * KNN_K * 3 * 4);
  float* featbf = (float*)alloc((size_t)B * V1 * 5120 * 4);   // 21 MB, max GEMM out
  float* fmA    = (float*)alloc((size_t)1048576 * 4);
  float* fmB    = (float*)alloc((size_t)1048576 * 4);
  float* gbuf   = (float*)alloc((size_t)B * 1024 * 4);
  float* verts1 = (float*)alloc((size_t)B * V1 * 3 * 4);
  float* verts2 = (float*)alloc((size_t)B * V2 * 3 * 4);

  auto gemmGrid = [](int R, int N) {
    return dim3((unsigned)CDIV(N / 16, 8), (unsigned)CDIV(R, 16));
  };

  // stage dir normalizations
  normalize_cols<<<CDIV(64, 128), 128, 0, stream>>>(dir0, sd0, 64);
  normalize_cols<<<CDIV(128, 128), 128, 0, stream>>>(dir1, sd1, 128);
  normalize_cols<<<CDIV(256, 128), 128, 0, stream>>>(dir2, sd2, 256);
  normalize_cols<<<CDIV(1024, 128), 128, 0, stream>>>(dir3, sd3, 1024);
  normalize_cols<<<CDIV(1024, 128), 128, 0, stream>>>(dir4, sd4, 1024);
  normalize_cols<<<CDIV(4096, 128), 128, 0, stream>>>(dir5, sd5, 4096);

  // --- knn0 + surface conv -> fm0(16)
  knn_topk<<<dim3(V0 / 16, B), 256, 0, stream>>>(verts0, idxb, V0, KNN_K + 1);
  nbr_dirs<<<CDIV(B * V0 * KNN_K, 256), 256, 0, stream>>>(verts0, idxb, dirn, B, V0, KNN_K);
  conv_theta<<<CDIV(B * V0 * 16, 256), 256, 0, stream>>>(nullptr, dirn, idxb, sd0, fmA, B, V0, 16, KNN_K);

  // --- conv1: feat1 = fm0@w1+b1, fm1(32)
  gemm_wmma_f16<<<gemmGrid(B * V0, 160), 256, 0, stream>>>(fmA, w1, b1, featbf, B * V0, 16, 160);
  conv_theta<<<CDIV(B * V0 * 32, 256), 256, 0, stream>>>(featbf, dirn, idxb, sd1, fmB, B, V0, 32, KNN_K);

  // --- pool1 (nbr=4) -> verts1, fm1p(32)
  knn_topk<<<dim3(V0 / 16, B), 256, 0, stream>>>(verts0, idxb, V0, 5);
  pool_verts<<<CDIV(B * V1 * 3, 256), 256, 0, stream>>>(verts0, verts1, V0, V1, B);
  pool_max<<<CDIV(B * V1 * 32, 256), 256, 0, stream>>>(fmB, idxb, fmA, B, V0, V1, 32, 4);

  // --- knn1 + conv2: fm2(64)
  knn_topk<<<dim3(V1 / 16, B), 256, 0, stream>>>(verts1, idxb, V1, KNN_K + 1);
  nbr_dirs<<<CDIV(B * V1 * KNN_K, 256), 256, 0, stream>>>(verts1, idxb, dirn, B, V1, KNN_K);
  gemm_wmma_f16<<<gemmGrid(B * V1, 320), 256, 0, stream>>>(fmA, w2, b2, featbf, B * V1, 32, 320);
  conv_theta<<<CDIV(B * V1 * 64, 256), 256, 0, stream>>>(featbf, dirn, idxb, sd2, fmB, B, V1, 64, KNN_K);

  // --- global-max concat -> fm2c(128)
  colmax<<<CDIV(B * 64, 128), 128, 0, stream>>>(fmB, gbuf, B, V1, 64);
  concat_g<<<CDIV(B * V1 * 128, 256), 256, 0, stream>>>(fmB, gbuf, fmA, B, V1, 64);

  // --- conv3: fm3(256)
  gemm_wmma_f16<<<gemmGrid(B * V1, 1280), 256, 0, stream>>>(fmA, w3, b3, featbf, B * V1, 128, 1280);
  conv_theta<<<CDIV(B * V1 * 256, 256), 256, 0, stream>>>(featbf, dirn, idxb, sd3, fmB, B, V1, 256, KNN_K);

  // --- pool2 -> verts2, fm3p(256)
  knn_topk<<<dim3(V1 / 16, B), 256, 0, stream>>>(verts1, idxb, V1, 5);
  pool_verts<<<CDIV(B * V2 * 3, 256), 256, 0, stream>>>(verts1, verts2, V1, V2, B);
  pool_max<<<CDIV(B * V2 * 256, 256), 256, 0, stream>>>(fmB, idxb, fmA, B, V1, V2, 256, 4);

  // --- knn2 + conv4: fm4(256)
  knn_topk<<<dim3(V2 / 16, B), 256, 0, stream>>>(verts2, idxb, V2, KNN_K + 1);
  nbr_dirs<<<CDIV(B * V2 * KNN_K, 256), 256, 0, stream>>>(verts2, idxb, dirn, B, V2, KNN_K);
  gemm_wmma_f16<<<gemmGrid(B * V2, 1280), 256, 0, stream>>>(fmA, w4, b4, featbf, B * V2, 256, 1280);
  conv_theta<<<CDIV(B * V2 * 256, 256), 256, 0, stream>>>(featbf, dirn, idxb, sd4, fmB, B, V2, 256, KNN_K);

  // --- global-max concat -> fm4c(512)
  colmax<<<CDIV(B * 256, 128), 128, 0, stream>>>(fmB, gbuf, B, V2, 256);
  concat_g<<<CDIV(B * V2 * 512, 256), 256, 0, stream>>>(fmB, gbuf, fmA, B, V2, 256);

  // --- conv5: fm5(1024)
  gemm_wmma_f16<<<gemmGrid(B * V2, 5120), 256, 0, stream>>>(fmA, w5, b5, featbf, B * V2, 512, 5120);
  conv_theta<<<CDIV(B * V2 * 1024, 256), 256, 0, stream>>>(featbf, dirn, idxb, sd5, fmB, B, V2, 1024, KNN_K);

  // --- global feature
  colmax<<<CDIV(B * 1024, 128), 128, 0, stream>>>(fmB, gbuf, B, V2, 1024);

  // --- heads
  float* hc = featbf;
  float* hd = featbf + B * 256;
  gemm_wmma_f16<<<gemmGrid(B, 256), 256, 0, stream>>>(gbuf, lc1w, lc1b, hc, B, 1024, 256);
  bn_relu<<<CDIV(B * 256, 256), 256, 0, stream>>>(hc, bncg, bncb, B, 256);
  head_out<<<1, 32, 0, stream>>>(hc, lc2w, lc2b, (float*)d_out, B, 256, 3);

  gemm_wmma_f16<<<gemmGrid(B, 256), 256, 0, stream>>>(gbuf, ld1w, ld1b, hd, B, 1024, 256);
  bn_relu<<<CDIV(B * 256, 256), 256, 0, stream>>>(hd, bndg, bndb, B, 256);
  head_out<<<1, 32, 0, stream>>>(hd, ld2w, ld2b, (float*)d_out + 12, B, 256, 2);
}
